// FastAttention_56367150793415
// MI455X (gfx1250) — matrix operand order repaired and checked by reading, not compiled, not run on hardware
//
#include <hip/hip_runtime.h>
#include <hip/hip_bf16.h>
#include <cstdint>

typedef __bf16 bf16_t;
typedef bf16_t v16bf __attribute__((ext_vector_type(16)));
typedef float  v8f   __attribute__((ext_vector_type(8)));

// ---------------------------------------------------------------------------
// WMMA wrapper: D = A(16x32 bf16) x B(32x16 bf16) + C(16x16 f32)
// ---------------------------------------------------------------------------
__device__ __forceinline__ v8f wmma_bf16(v16bf a, v16bf b, v8f c) {
  return __builtin_amdgcn_wmma_f32_16x16x32_bf16(
      /*neg_a=*/false, a, /*neg_b=*/false, b,
      /*c_mod=*/(short)0, c, /*reuse_a=*/false, /*reuse_b=*/false);
}

// ---------------------------------------------------------------------------
// Async global -> LDS copy (CDNA5, ASYNCcnt-tracked, bypasses VGPRs).
// GV mode: 64-bit per-lane global address, per-lane 32-bit LDS address.
// ---------------------------------------------------------------------------
__device__ __forceinline__ unsigned lds_addr32(const void* p) {
  // LDS aperture: generic addr[31:0] == LDS byte offset (ISA 10.2)
  return (unsigned)(uintptr_t)p;
}

__device__ __forceinline__ void async_copy_b128(unsigned lds, const void* gptr) {
  asm volatile("global_load_async_to_lds_b128 %0, %1, off"
               :: "v"(lds), "v"(gptr)
               : "memory");
}

__device__ __forceinline__ void wait_async0() {
  asm volatile("s_wait_asynccnt 0x0" ::: "memory");
}

// ---------------------------------------------------------------------------
// Fragment loaders (A/B: contiguous k = 16*(lane/16) + s; C/D: ISA 7.12.2 layout; wave32)
// ---------------------------------------------------------------------------
__device__ __forceinline__ v16bf frag_a_rm_bf(const bf16_t* p, int ld,
                                              int row0, int k0, int lane) {
  int m = lane & 15, h = (lane >> 4) * 16;
  const bf16_t* q = p + (long long)(row0 + m) * ld + k0;
  v16bf a;
#pragma unroll
  for (int i = 0; i < 8; ++i) {
    int kb = 2 * i + h;
    a[2 * i]     = q[kb];
    a[2 * i + 1] = q[kb + 1];
  }
  return a;
}

__device__ __forceinline__ v16bf frag_a_rm_f32(const float* p, int ld,
                                               int row0, int k0, int lane) {
  int m = lane & 15, h = (lane >> 4) * 16;
  const float* q = p + (long long)(row0 + m) * ld + k0;
  v16bf a;
#pragma unroll
  for (int i = 0; i < 8; ++i) {
    int kb = 2 * i + h;
    a[2 * i]     = (bf16_t)q[kb];
    a[2 * i + 1] = (bf16_t)q[kb + 1];
  }
  return a;
}

// Transposed A: logical A[row][k] = p[k*ld + row]
__device__ __forceinline__ v16bf frag_a_T_bf(const bf16_t* p, int ld,
                                             int row0, int k0, int lane) {
  int m = lane & 15, h = (lane >> 4) * 16;
  const bf16_t* q = p + row0 + m;
  v16bf a;
#pragma unroll
  for (int i = 0; i < 8; ++i) {
    int kb = 2 * i + h + k0;
    a[2 * i]     = q[(long long)kb * ld];
    a[2 * i + 1] = q[(long long)(kb + 1) * ld];
  }
  return a;
}

// B stored column-major in LDS: p[col*ld + k]
__device__ __forceinline__ v16bf frag_b_cm_bf(const bf16_t* p, int ld,
                                              int col0, int k0, int lane) {
  int n = lane & 15, h = (lane >> 4) * 16;
  const bf16_t* q = p + (long long)(col0 + n) * ld + k0 + h;
  v16bf b;
#pragma unroll
  for (int i = 0; i < 8; ++i) { b[2 * i] = q[2 * i]; b[2 * i + 1] = q[2 * i + 1]; }
  return b;
}

// B from row-major [k][n] storage
__device__ __forceinline__ v16bf frag_b_rm_bf(const bf16_t* p, int ld,
                                              int k0, int col0, int lane) {
  int n = lane & 15, h = (lane >> 4) * 16;
  const bf16_t* q = p + (long long)(k0 + h) * ld + col0 + n;
  v16bf b;
#pragma unroll
  for (int i = 0; i < 8; ++i) {
    b[2 * i]     = q[(long long)(2 * i) * ld];
    b[2 * i + 1] = q[(long long)(2 * i + 1) * ld];
  }
  return b;
}

__device__ __forceinline__ v16bf frag_b_rm_f32(const float* p, int ld,
                                               int k0, int col0, int lane) {
  int n = lane & 15, h = (lane >> 4) * 16;
  const float* q = p + (long long)(k0 + h) * ld + col0 + n;
  v16bf b;
#pragma unroll
  for (int i = 0; i < 8; ++i) {
    b[2 * i]     = (bf16_t)q[(long long)(2 * i) * ld];
    b[2 * i + 1] = (bf16_t)q[(long long)(2 * i + 1) * ld];
  }
  return b;
}

// C/D 16x16 f32: VGPR r, lane l -> M = r + 8*(l>=16), N = l&15
__device__ __forceinline__ v8f frag_c_rm_f32(const float* p, int ld,
                                             int row0, int col0, int lane) {
  int n = lane & 15, h = (lane >> 4) * 8;
  v8f c;
#pragma unroll
  for (int r = 0; r < 8; ++r)
    c[r] = p[(long long)(row0 + h + r) * ld + col0 + n];
  return c;
}

__device__ __forceinline__ void store_c_rm_f32(float* p, int ld, int row0,
                                               int col0, v8f c, int lane) {
  int n = lane & 15, h = (lane >> 4) * 8;
#pragma unroll
  for (int r = 0; r < 8; ++r)
    p[(long long)(row0 + h + r) * ld + col0 + n] = c[r];
}

// ---------------------------------------------------------------------------
// fp32 -> bf16 cast
// ---------------------------------------------------------------------------
__global__ void cast_f32_to_bf16(const float* __restrict__ in,
                                 bf16_t* __restrict__ out, int n) {
  int i = blockIdx.x * blockDim.x + threadIdx.x;
  if (i < n) out[i] = (bf16_t)in[i];
}

// ---------------------------------------------------------------------------
// Generic bf16 WMMA GEMM with async double-buffered A staging.
// Block: 128 threads (4 waves). Block tile 256x64, wave tile 64x64 (4x4 WMMA).
// mode 0: fp32 out + bias.   mode 1: relu(x)*scale -> bf16 out.
// ---------------------------------------------------------------------------
__global__ __launch_bounds__(128)
void gemm_bf16_kernel(const bf16_t* __restrict__ A, int lda, long long Abatch,
                      const bf16_t* __restrict__ B, int ldb, long long Bbatch,
                      const float* __restrict__ bias,
                      float* __restrict__ Cf, bf16_t* __restrict__ Cb,
                      int ldc, long long Cbatch,
                      int K, int mode, float scale) {
  const int tid = threadIdx.x, w = tid >> 5, lane = tid & 31;
  const int row0 = blockIdx.x * 256;
  const int col0 = blockIdx.y * 64;
  const long long z = blockIdx.z;

  const bf16_t* Ag = A + z * Abatch;
  const bf16_t* Bg = B + z * Bbatch;

  __shared__ alignas(16) bf16_t As[2][256 * 40];  // row-major [256][32] pad 40
  __shared__ alignas(16) bf16_t Bs[2][64 * 40];   // col-major [64][32] pad 40

  // Async-stage a 256x32 A tile (each thread: 2 rows x 4 b128, no VGPR data)
  auto stage_A = [&](int buf, int kk) {
    int r = tid * 2;
    const bf16_t* s0 = Ag + (long long)(row0 + r) * lda + kk;
    const bf16_t* s1 = s0 + lda;
    unsigned l0 = lds_addr32(&As[buf][r * 40]);
    unsigned l1 = l0 + 80;
#pragma unroll
    for (int q = 0; q < 4; ++q) {
      async_copy_b128(l0 + q * 16, s0 + q * 8);
      async_copy_b128(l1 + q * 16, s1 + q * 8);
    }
  };
  // Sync-stage B transposed (element transpose; contiguous global reads)
  auto stage_B = [&](int buf, int kk) {
    int kr = tid >> 2;
    int n0 = (tid & 3) * 16;
    const bf16_t* bs = Bg + (long long)(kk + kr) * ldb + col0 + n0;
#pragma unroll
    for (int j = 0; j < 16; ++j) Bs[buf][(n0 + j) * 40 + kr] = bs[j];
  };

  const v8f zero8 = {0.f, 0.f, 0.f, 0.f, 0.f, 0.f, 0.f, 0.f};
  v8f acc[4][4];
#pragma unroll
  for (int mi = 0; mi < 4; ++mi)
#pragma unroll
    for (int ni = 0; ni < 4; ++ni) acc[mi][ni] = zero8;

  // Prologue: stage tile 0
  stage_A(0, 0);
  stage_B(0, 0);
  wait_async0();
  __syncthreads();

  const int nsteps = K >> 5;
  for (int s = 0; s < nsteps; ++s) {
    const int cur = s & 1;
    if (s + 1 < nsteps) {       // prefetch next tile into the other buffer
      stage_A(cur ^ 1, (s + 1) * 32);
      stage_B(cur ^ 1, (s + 1) * 32);
    }

    v16bf af[4];
#pragma unroll
    for (int mi = 0; mi < 4; ++mi)
      af[mi] = frag_a_rm_bf(As[cur], 40, w * 64 + mi * 16, 0, lane);
#pragma unroll
    for (int ni = 0; ni < 4; ++ni) {
      v16bf bb = frag_b_cm_bf(Bs[cur], 40, ni * 16, 0, lane);
#pragma unroll
      for (int mi = 0; mi < 4; ++mi)
        acc[mi][ni] = wmma_bf16(af[mi], bb, acc[mi][ni]);
    }

    wait_async0();              // next tile's async copies complete
    __syncthreads();            // all waves done reading cur before re-stage
  }

  // Epilogue
  {
    int n = lane & 15, hh = (lane >> 4) * 8;
#pragma unroll
    for (int mi = 0; mi < 4; ++mi)
#pragma unroll
      for (int ni = 0; ni < 4; ++ni)
#pragma unroll
        for (int r = 0; r < 8; ++r) {
          long long row = row0 + w * 64 + mi * 16 + hh + r;
          long long col = col0 + ni * 16 + n;
          float v = acc[mi][ni][r];
          if (mode == 0) {
            if (bias) v += bias[col];
            Cf[z * Cbatch + row * ldc + col] = v;
          } else {
            v = fmaxf(v, 0.f) * scale;
            Cb[z * Cbatch + row * ldc + col] = (bf16_t)v;
          }
        }
  }
}

// ---------------------------------------------------------------------------
// Chunked causal linear attention. One block per head, 128 threads (4 waves).
// Chunk S=64; all GEMMs 64x64x64 via WMMA; chunk loads via async global->LDS.
// ---------------------------------------------------------------------------
#define ALQ 72  // padded bf16 row length for 64-wide tiles

__global__ __launch_bounds__(128)
void lin_attn_kernel(const bf16_t* __restrict__ qprime,  // [H][T][64]
                     const bf16_t* __restrict__ kprime,  // [H][T][64]
                     const bf16_t* __restrict__ qkv_bf,  // [T][3072]
                     bf16_t* __restrict__ attn_out,      // [T][1024]
                     int T) {
  const int h = blockIdx.x;
  const int tid = threadIdx.x;
  const int w = tid >> 5, lane = tid & 31;

  __shared__ alignas(16) bf16_t Qs[64 * ALQ];   // [i][m]
  __shared__ alignas(16) bf16_t Ks[64 * ALQ];   // [j][m]
  __shared__ alignas(16) bf16_t Vs[64 * ALQ];   // [j][d]
  __shared__ float  Scf[64 * 65];               // masked scores [i][j]
  __shared__ float  Sf[64 * 65];                // state [m][d] fp32
  __shared__ float  ksum[64];
  __shared__ float  nrm[64];

  for (int idx = tid; idx < 64 * 65; idx += 128) Sf[idx] = 0.f;
  if (tid < 64) ksum[tid] = 0.f;
  __syncthreads();

  const bf16_t* qh = qprime + (long long)h * T * 64;
  const bf16_t* kh = kprime + (long long)h * T * 64;
  const bf16_t* vh = qkv_bf + 2048 + h * 64;  // V block at col 2*C

  const v8f zero8 = {0.f, 0.f, 0.f, 0.f, 0.f, 0.f, 0.f, 0.f};
  const int nchunks = T / 64;

  for (int c = 0; c < nchunks; ++c) {
    const int t0 = c * 64;
    // ---- async-load chunk: each thread half a row of Q', K', V ----
    {
      int r = tid >> 1, hc = (tid & 1) * 32;
      const bf16_t* qsrc = qh + (long long)(t0 + r) * 64 + hc;
      const bf16_t* ksrc = kh + (long long)(t0 + r) * 64 + hc;
      const bf16_t* vsrc = vh + (long long)(t0 + r) * 3072 + hc;
      unsigned lq = lds_addr32(&Qs[r * ALQ + hc]);
      unsigned lk = lds_addr32(&Ks[r * ALQ + hc]);
      unsigned lv = lds_addr32(&Vs[r * ALQ + hc]);
#pragma unroll
      for (int q = 0; q < 4; ++q) {
        async_copy_b128(lq + q * 16, qsrc + q * 8);
        async_copy_b128(lk + q * 16, ksrc + q * 8);
        async_copy_b128(lv + q * 16, vsrc + q * 8);
      }
      wait_async0();
    }
    __syncthreads();

    // ---- scores = Q' @ K'^T : wave w -> rows 16w..16w+15, all 64 cols ----
    v8f sc[4] = {zero8, zero8, zero8, zero8};
    for (int kstep = 0; kstep < 64; kstep += 32) {
      v16bf aq = frag_a_rm_bf(Qs, ALQ, w * 16, kstep, lane);
#pragma unroll
      for (int ni = 0; ni < 4; ++ni) {
        v16bf bk = frag_b_cm_bf(Ks, ALQ, ni * 16, kstep, lane);
        sc[ni] = wmma_bf16(aq, bk, sc[ni]);
      }
    }
    // causal mask, stage fp32
    {
      int n = lane & 15, hh = (lane >> 4) * 8;
#pragma unroll
      for (int ni = 0; ni < 4; ++ni)
#pragma unroll
        for (int r = 0; r < 8; ++r) {
          int i = w * 16 + hh + r, j = ni * 16 + n;
          Scf[i * 65 + j] = (j <= i) ? sc[ni][r] : 0.f;
        }
    }
    __syncthreads();

    // ---- norm: q'.ksum_prev + rowsum(masked scores) ----
    if (tid < 64) {
      float s = 0.f;
      for (int m = 0; m < 64; ++m) s += (float)Qs[tid * ALQ + m] * ksum[m];
      for (int j = 0; j < 64; ++j) s += Scf[tid * 65 + j];
      nrm[tid] = s + 1e-6f;
    }
    __syncthreads();

    // ---- ctx = Q'@state + masked_scores@V ----
    v8f acc[4] = {zero8, zero8, zero8, zero8};
    for (int kstep = 0; kstep < 64; kstep += 32) {
      v16bf aq = frag_a_rm_bf(Qs, ALQ, w * 16, kstep, lane);
      v16bf as = frag_a_rm_f32(Scf, 65, w * 16, kstep, lane);
#pragma unroll
      for (int ni = 0; ni < 4; ++ni) {
        v16bf bst = frag_b_rm_f32(Sf, 65, kstep, ni * 16, lane);   // state [m][d]
        v16bf bv  = frag_b_rm_bf(Vs, ALQ, kstep, ni * 16, lane);   // V [j][d]
        acc[ni] = wmma_bf16(aq, bst, acc[ni]);
        acc[ni] = wmma_bf16(as, bv, acc[ni]);
      }
    }
    // divide by norm, write bf16 at head column block
    {
      int n = lane & 15, hh = (lane >> 4) * 8;
#pragma unroll
      for (int ni = 0; ni < 4; ++ni)
#pragma unroll
        for (int r = 0; r < 8; ++r) {
          int i = w * 16 + hh + r, d = ni * 16 + n;
          float ov = acc[ni][r] / nrm[i];
          attn_out[(long long)(t0 + i) * 1024 + h * 64 + d] = (bf16_t)ov;
        }
    }
    // ksum += colsum(K')  (next reader is next chunk's norm pass)
    if (tid < 64) {
      float s = ksum[tid];
      for (int j = 0; j < 64; ++j) s += (float)Ks[j * ALQ + tid];
      ksum[tid] = s;
    }
    __syncthreads();  // all Sf reads (ctx GEMM) complete before update

    // ---- state += K'^T @ V : wave w owns state rows 16w..16w+15 ----
    v8f st[4];
#pragma unroll
    for (int ni = 0; ni < 4; ++ni)
      st[ni] = frag_c_rm_f32(Sf, 65, w * 16, ni * 16, lane);
    for (int kstep = 0; kstep < 64; kstep += 32) {
      v16bf akt = frag_a_T_bf(Ks, ALQ, w * 16, kstep, lane);  // A[m][j]=Ks[j][m]
#pragma unroll
      for (int ni = 0; ni < 4; ++ni) {
        v16bf bv = frag_b_rm_bf(Vs, ALQ, kstep, ni * 16, lane);
        st[ni] = wmma_bf16(akt, bv, st[ni]);
      }
    }
#pragma unroll
    for (int ni = 0; ni < 4; ++ni)
      store_c_rm_f32(Sf, 65, w * 16, ni * 16, st[ni], lane);
    __syncthreads();
  }
}

// ---------------------------------------------------------------------------
// Host launch
// ---------------------------------------------------------------------------
extern "C" void kernel_launch(void* const* d_in, const int* in_sizes, int n_in,
                              void* d_out, int out_size, void* d_ws, size_t ws_size,
                              hipStream_t stream) {
  (void)in_sizes; (void)n_in; (void)out_size; (void)ws_size;
  const float* x     = (const float*)d_in[0];
  const float* w_qkv = (const float*)d_in[1];
  const float* b_qkv = (const float*)d_in[2];
  const float* w_out = (const float*)d_in[3];
  const float* b_out = (const float*)d_in[4];
  const float* proj  = (const float*)d_in[5];

  const int T = 2048, C = 1024, H = 16, Dh = 64, Mf = 64, C3 = 3072;

  char* ws = (char*)d_ws;
  size_t off = 0;
  auto carve = [&](size_t bytes) -> void* {
    void* p = ws + off;
    off += (bytes + 255) & ~(size_t)255;
    return p;
  };

  bf16_t* x_bf    = (bf16_t*)carve((size_t)T * C * 2);
  bf16_t* wqkv_bf = (bf16_t*)carve((size_t)C * C3 * 2);
  bf16_t* wout_bf = (bf16_t*)carve((size_t)C * C * 2);
  bf16_t* proj_bf = (bf16_t*)carve((size_t)H * Dh * Mf * 2);
  float*  qkv_f   = (float*)carve((size_t)T * C3 * 4);
  bf16_t* qkv_bf  = (bf16_t*)carve((size_t)T * C3 * 2);
  bf16_t* qprime  = (bf16_t*)carve((size_t)H * T * Mf * 2);
  bf16_t* kprime  = (bf16_t*)carve((size_t)H * T * Mf * 2);
  bf16_t* attn_bf = (bf16_t*)carve((size_t)T * C * 2);

  const int CT = 256;
  cast_f32_to_bf16<<<(T * C + CT - 1) / CT, CT, 0, stream>>>(x, x_bf, T * C);
  cast_f32_to_bf16<<<(C * C3 + CT - 1) / CT, CT, 0, stream>>>(w_qkv, wqkv_bf, C * C3);
  cast_f32_to_bf16<<<(C * C + CT - 1) / CT, CT, 0, stream>>>(w_out, wout_bf, C * C);
  cast_f32_to_bf16<<<(H * Dh * Mf + CT - 1) / CT, CT, 0, stream>>>(proj, proj_bf, H * Dh * Mf);

  dim3 blk(128);

  // QKV projection: [T,C] @ [C,3C] + b_qkv -> fp32
  gemm_bf16_kernel<<<dim3(T / 256, C3 / 64, 1), blk, 0, stream>>>(
      x_bf, C, 0, wqkv_bf, C3, 0, b_qkv, qkv_f, (bf16_t*)nullptr, C3, 0,
      C, 0, 1.0f);

  cast_f32_to_bf16<<<(T * C3 + CT - 1) / CT, CT, 0, stream>>>(qkv_f, qkv_bf, T * C3);

  // Feature maps: per head [T,64] @ [64,64], relu * 1/sqrt(64) -> bf16
  gemm_bf16_kernel<<<dim3(T / 256, 1, H), blk, 0, stream>>>(
      qkv_bf, C3, 64, proj_bf, Mf, (long long)Dh * Mf, nullptr,
      (float*)nullptr, qprime, Mf, (long long)T * Mf, Dh, 1, 0.125f);
  gemm_bf16_kernel<<<dim3(T / 256, 1, H), blk, 0, stream>>>(
      qkv_bf + C, C3, 64, proj_bf, Mf, (long long)Dh * Mf, nullptr,
      (float*)nullptr, kprime, Mf, (long long)T * Mf, Dh, 1, 0.125f);

  // Causal linear attention scan (chunked, WMMA)
  lin_attn_kernel<<<dim3(H), dim3(128), 0, stream>>>(qprime, kprime, qkv_bf,
                                                     attn_bf, T);

  // Output projection: [T,C] @ [C,C] + b_out -> fp32 d_out
  gemm_bf16_kernel<<<dim3(T / 256, C / 64, 1), blk, 0, stream>>>(
      attn_bf, C, 0, wout_bf, C, 0, b_out, (float*)d_out, (bf16_t*)nullptr,
      C, 0, C, 0, 1.0f);
}
